// RefAttnProcessor_V2_15324443312824
// MI455X (gfx1250) — matrix operand order repaired and checked by reading, not compiled, not run on hardware
//
#include <hip/hip_runtime.h>
#include <stdint.h>

typedef _Float16 f16;
typedef __attribute__((ext_vector_type(16))) _Float16 v16h;
typedef __attribute__((ext_vector_type(8)))  float    v8f;

#define HIDDEN 3072
#define DD 128
#define NH 24
#define LK 1024
#define LQ 4096

// attn LDS layout (dynamic shared segment)
#define KS_BYTES (64 * 128 * 2)        // K tile  [64 keys][128 d] f16   = 16384
#define VS_STRIDE 80                   // halfs per row (160B, 16B-aligned)
#define VS_BYTES (128 * VS_STRIDE * 2) // V tile^T [128 d][80]           = 20480
#define PS_BYTES (4 * 2 * 32 * 16 * 2) // P frags [wave][grp][lane][16]  = 8192
#define SMEM_BYTES (KS_BYTES + VS_BYTES + PS_BYTES)

static __device__ __forceinline__ v8f wmma32f16(v16h a, v16h b, v8f c) {
  // D(f32 16x16) = A(f16 16x32) * B(f16 32x16) + C
  return __builtin_amdgcn_wmma_f32_16x16x32_f16(false, a, false, b, (short)0, c,
                                                false, false);
}

// ---------------------------------------------------------------------------
// Kernel 1: K/V projection GEMM (WMMA) with fused RMSNorm + RoPE epilogue (K).
// Workgroup = 8 waves; tile = 16 rows x 128 cols (= one head's full D).
// grid = (64 row-tiles, 24 heads, 2 {K,V})
// ---------------------------------------------------------------------------
__global__ __launch_bounds__(256) void proj_kernel(
    const float* __restrict__ latent, const float* __restrict__ Wk,
    const float* __restrict__ Wv, const float* __restrict__ knw,
    const float* __restrict__ fcos, const float* __restrict__ fsin,
    f16* __restrict__ Kf, f16* __restrict__ Vf)
{
  const int mt   = blockIdx.x;
  const int h    = blockIdx.y;
  const bool isK = (blockIdx.z == 0);
  const float* W = isK ? Wk : Wv;
  const int lane = threadIdx.x & 31;
  const int w    = threadIdx.x >> 5;
  const int nb   = h * DD + w * 16;
  const int mrow = mt * 16 + (lane & 15);      // A-frag: lane = M row
  const int ncol = nb + (lane & 15);           // B-frag: lane = N col
  const int ak0  = (lane & 16) ? 8 : 0;        // A-frag K grouping (ISA table)
  const int bk0  = (lane & 16) ? 8 : 0;        // B-frag K grouping (ISA table)

  v8f acc = {};
  for (int k = 0; k < HIDDEN; k += 32) {
    v16h a, b;
    const float* ap = latent + (long)mrow * HIDDEN + k + ak0;
#pragma unroll
    for (int i = 0; i < 8; ++i) { a[i] = (f16)ap[i]; a[8 + i] = (f16)ap[16 + i]; }
    const float* bp = W + (long)ncol * HIDDEN + k + bk0;   // B[kd][n] = W[n][k+kd]
#pragma unroll
    for (int i = 0; i < 8; ++i) { b[i] = (f16)bp[i]; b[8 + i] = (f16)bp[16 + i]; }
    acc = wmma32f16(a, b, acc);
  }

  if (!isK) {
#pragma unroll
    for (int v = 0; v < 8; ++v) {
      int r = mt * 16 + ((lane & 16) ? v + 8 : v);
      Vf[(long)r * HIDDEN + ncol] = (f16)acc[v];
    }
    return;
  }

  __shared__ float Cs[16][132];
  __shared__ float rstdS[16];
#pragma unroll
  for (int v = 0; v < 8; ++v) {
    int r = (lane & 16) ? v + 8 : v;
    Cs[r][w * 16 + (lane & 15)] = acc[v];
  }
  __syncthreads();
  if (threadIdx.x < 16) {
    float ss = 0.f;
    for (int c = 0; c < DD; ++c) { float x = Cs[threadIdx.x][c]; ss += x * x; }
    rstdS[threadIdx.x] = rsqrtf(ss * (1.f / DD) + 1e-6f);
  }
  __syncthreads();
  for (int p = threadIdx.x; p < 16 * 64; p += 256) {
    int r  = p >> 6;
    int dp = (p & 63) * 2;
    int l  = mt * 16 + r;
    float rs = rstdS[r];
    float x1 = Cs[r][dp]     * rs * knw[dp];
    float x2 = Cs[r][dp + 1] * rs * knw[dp + 1];
    float c1 = fcos[l * DD + dp],     s1 = fsin[l * DD + dp];
    float c2 = fcos[l * DD + dp + 1], s2 = fsin[l * DD + dp + 1];
    Kf[(long)l * HIDDEN + h * DD + dp]     = (f16)(x1 * c1 - x2 * s1);
    Kf[(long)l * HIDDEN + h * DD + dp + 1] = (f16)(x2 * c2 + x1 * s2);
  }
}

// ---------------------------------------------------------------------------
// Kernel 2: RoPE(q) -> f16, position = row mod 1024.
// ---------------------------------------------------------------------------
__global__ __launch_bounds__(256) void rope_q_kernel(
    const float* __restrict__ q, const float* __restrict__ fcos,
    const float* __restrict__ fsin, f16* __restrict__ Qf)
{
  long p = (long)blockIdx.x * blockDim.x + threadIdx.x;
  if (p >= (long)LQ * (HIDDEN / 2)) return;
  int colp = (int)(p % (HIDDEN / 2));
  int qr   = (int)(p / (HIDDEN / 2));
  int col  = colp * 2;
  int dp   = col & (DD - 1);
  int l    = qr & (LK - 1);
  float x1 = q[(long)qr * HIDDEN + col];
  float x2 = q[(long)qr * HIDDEN + col + 1];
  float c1 = fcos[l * DD + dp],     s1 = fsin[l * DD + dp];
  float c2 = fcos[l * DD + dp + 1], s2 = fsin[l * DD + dp + 1];
  Qf[(long)qr * HIDDEN + col]     = (f16)(x1 * c1 - x2 * s1);
  Qf[(long)qr * HIDDEN + col + 1] = (f16)(x2 * c2 + x1 * s2);
}

// ---------------------------------------------------------------------------
// Kernel 3: flash attention. WG = 4 waves x 16 q-rows = 64 q rows.
// 64-key tiles; K tile staged via global_load_async_to_lds_b128 (ASYNCcnt),
// V tile staged transposed; P written directly in A-fragment order.
// grid = (4096/64, 24 heads)
// ---------------------------------------------------------------------------
__global__ __launch_bounds__(128) void attn_kernel(
    const f16* __restrict__ Qf, const f16* __restrict__ Kf,
    const f16* __restrict__ Vf, float* __restrict__ out)
{
  extern __shared__ __align__(16) char smem[];
  f16* Ks = (f16*)smem;                           // [64][128]
  f16* Vs = (f16*)(smem + KS_BYTES);              // [128][VS_STRIDE]
  f16* Ps = (f16*)(smem + KS_BYTES + VS_BYTES);   // [4][2][32][16]

  const int qt   = blockIdx.x;
  const int h    = blockIdx.y;
  const int lane = threadIdx.x & 31;
  const int w    = threadIdx.x >> 5;
  const int qrow = qt * 64 + w * 16 + (lane & 15);
  const int ak0  = (lane & 16) ? 8 : 0;
  const int bk0  = (lane & 16) ? 8 : 0;
  const int ncol = lane & 15;

  // This wave's Q fragments: 4 K-slices of 32 over d=0..127
  v16h aq[4];
#pragma unroll
  for (int s = 0; s < 4; ++s) {
    const f16* qp = Qf + (long)qrow * HIDDEN + h * DD + s * 32 + ak0;
#pragma unroll
    for (int i = 0; i < 8; ++i) { aq[s][i] = qp[i]; aq[s][8 + i] = qp[16 + i]; }
  }

  float mi[8], li[8];
  v8f acc[8];
  v8f zero = {};
#pragma unroll
  for (int v = 0; v < 8; ++v) { mi[v] = -1e30f; li[v] = 0.f; acc[v] = zero; }

  const float scale = 0.08838834764831845f;       // 1/sqrt(128)
  const uint32_t ksBase = (uint32_t)(uintptr_t)Ks; // LDS offset of K tile

  for (int j = 0; j < LK; j += 64) {
    __syncthreads();

    // --- async copy K tile (64 rows x 256B) to LDS: 1024 x 16B chunks ---
    for (int c = threadIdx.x; c < 1024; c += 128) {
      int row = c >> 4;
      int o   = (c & 15) * 16;                     // byte offset within row
      uint64_t ga = (uint64_t)(uintptr_t)(Kf + (size_t)(j + row) * HIDDEN + h * DD) + o;
      uint32_t la = ksBase + row * 256 + o;
      asm volatile("global_load_async_to_lds_b128 %0, %1, off"
                   :: "v"(la), "v"(ga) : "memory");
    }
    // --- stage V tile transposed: Vs[d][key] ---
    for (int e = threadIdx.x; e < 64 * DD; e += 128) {
      int kk = e >> 7;
      int dd = e & (DD - 1);
      Vs[dd * VS_STRIDE + kk] = Vf[(size_t)(j + kk) * HIDDEN + h * DD + dd];
    }
    asm volatile("s_wait_asynccnt 0" ::: "memory");
    __syncthreads();

    // --- S = Q K^T over 64 keys: four 16x16 tiles ---
    v8f c4[4] = {zero, zero, zero, zero};
#pragma unroll
    for (int s = 0; s < 4; ++s) {
#pragma unroll
      for (int kg = 0; kg < 4; ++kg) {
        const f16* kp = Ks + (kg * 16 + ncol) * 128 + s * 32 + bk0;
        v16h b;
#pragma unroll
        for (int i = 0; i < 8; ++i) { b[i] = kp[i]; b[8 + i] = kp[16 + i]; }
        c4[kg] = wmma32f16(aq[s], b, c4[kg]);
      }
    }

    // --- online softmax (row = C-layout VGPR idx, +8 on high lane half) ---
#pragma unroll
    for (int v = 0; v < 8; ++v) {
      float x0 = c4[0][v] * scale, x1 = c4[1][v] * scale;
      float x2 = c4[2][v] * scale, x3 = c4[3][v] * scale;
      float mx = fmaxf(fmaxf(x0, x1), fmaxf(x2, x3));
      mx = fmaxf(mx, __shfl_xor(mx, 1, 32));
      mx = fmaxf(mx, __shfl_xor(mx, 2, 32));
      mx = fmaxf(mx, __shfl_xor(mx, 4, 32));
      mx = fmaxf(mx, __shfl_xor(mx, 8, 32));
      float mnew = fmaxf(mi[v], mx);
      float p0 = __expf(x0 - mnew), p1 = __expf(x1 - mnew);
      float p2 = __expf(x2 - mnew), p3 = __expf(x3 - mnew);
      float rs = (p0 + p1) + (p2 + p3);
      rs += __shfl_xor(rs, 1, 32);
      rs += __shfl_xor(rs, 2, 32);
      rs += __shfl_xor(rs, 4, 32);
      rs += __shfl_xor(rs, 8, 32);
      float al = __expf(mi[v] - mnew);
      li[v] = li[v] * al + rs;
      mi[v] = mnew;

      // Write P directly in A-fragment order:
      //  element (row r, key c): lane' = r + 16*((c>>3)&1),
      //                          half  = (c&7) + 8*((c>>4)&1)
      int r  = (lane & 16) ? v + 8 : v;
      int lp = r + ((ncol & 8) ? 16 : 0);
      int hb = ncol & 7;
      f16* pw0 = Ps + ((w * 2 + 0) * 32 + lp) * 16;  // keys 0..31  (c=ncol / 16+ncol)
      f16* pw1 = Ps + ((w * 2 + 1) * 32 + lp) * 16;  // keys 32..63
      pw0[hb]     = (f16)p0;
      pw0[hb + 8] = (f16)p1;
      pw1[hb]     = (f16)p2;
      pw1[hb + 8] = (f16)p3;
#pragma unroll
      for (int f = 0; f < 8; ++f) acc[f][v] *= al;
    }

    // wave-local LDS RAW fence before reading fragments written by other lanes
    asm volatile("s_wait_dscnt 0" ::: "memory");

    v16h ap[2];
#pragma unroll
    for (int g = 0; g < 2; ++g) {
      const f16* pr = Ps + ((w * 2 + g) * 32 + lane) * 16;
#pragma unroll
      for (int i = 0; i < 16; ++i) ap[g][i] = pr[i];
    }

    // --- out += P(16x64) * V(64x128) ---
#pragma unroll
    for (int f = 0; f < 8; ++f) {
#pragma unroll
      for (int g = 0; g < 2; ++g) {
        const f16* vp = Vs + (f * 16 + ncol) * VS_STRIDE + g * 32 + bk0;
        v16h bv;
#pragma unroll
        for (int i = 0; i < 8; ++i) { bv[i] = vp[i]; bv[8 + i] = vp[16 + i]; }
        acc[f] = wmma32f16(ap[g], bv, acc[f]);
      }
    }
  }

  // --- finalize ---
#pragma unroll
  for (int f = 0; f < 8; ++f) {
#pragma unroll
    for (int v = 0; v < 8; ++v) {
      int r   = (lane & 16) ? v + 8 : v;
      int row = qt * 64 + w * 16 + r;
      out[(long)row * HIDDEN + h * DD + f * 16 + ncol] = acc[f][v] / li[v];
    }
  }
}

extern "C" void kernel_launch(void* const* d_in, const int* in_sizes, int n_in,
                              void* d_out, int out_size, void* d_ws, size_t ws_size,
                              hipStream_t stream)
{
  const float* query  = (const float*)d_in[0];
  const float* latent = (const float*)d_in[1];
  const float* Wk     = (const float*)d_in[2];
  const float* Wv     = (const float*)d_in[3];
  const float* knw    = (const float*)d_in[4];
  const float* fcos   = (const float*)d_in[5];
  const float* fsin   = (const float*)d_in[6];
  float* out = (float*)d_out;

  // Workspace (f16): Q[4096x3072] | K[1024x3072] | V[1024x3072] ~ 38 MB
  f16* Qf = (f16*)d_ws;
  f16* Kf = Qf + (size_t)LQ * HIDDEN;
  f16* Vf = Kf + (size_t)LK * HIDDEN;

  proj_kernel<<<dim3(64, NH, 2), 256, 0, stream>>>(latent, Wk, Wv, knw, fcos, fsin,
                                                   Kf, Vf);
  long pairs = (long)LQ * (HIDDEN / 2);
  rope_q_kernel<<<dim3((unsigned)((pairs + 255) / 256)), 256, 0, stream>>>(
      query, fcos, fsin, Qf);
  attn_kernel<<<dim3(LQ / 64, NH), 128, SMEM_BYTES, stream>>>(Qf, Kf, Vf, out);
}
